// EdgeReasoner_72164040508152
// MI455X (gfx1250) — matrix operand order repaired and checked by reading, not compiled, not run on hardware
//
#include <hip/hip_runtime.h>

// ---------------------------------------------------------------------------
// EdgeReasoner for MI455X (gfx1250, wave32, WMMA).
//   out = relu(BN(concat(x[src],x[dst]) @ W1^T)) @ W2^T + b2
// b1 is dropped: BatchNorm's mean subtraction absorbs it exactly.
// Pass 1: bf16 WMMA GEMM + per-feature sum/sumsq (global f32 atomics).
// Pass 2: recompute GEMM, fuse BN+ReLU+tiny projection, write [E,2] f32.
// Round-1: defeat LICM hoisting of LDS B-frag loads (512-VGPR scratch spill).
// Round-2: 3-deep software pipeline on B fragments so ds_load latency is
//          covered by in-flight WMMAs instead of s_wait_dscnt 0x0 stalls.
// ---------------------------------------------------------------------------

typedef __attribute__((ext_vector_type(16))) __bf16 v16bf;
typedef __attribute__((ext_vector_type(8)))  float  v8f;
typedef __attribute__((ext_vector_type(4)))  float  v4f;

#define FEAT 128          // D
#define FEAT2 256         // 2D
#define TILE_E 16         // edges per wave-tile
#define BLK_THREADS 256   // 8 waves / block
#define MAX_BLOCKS 2048

// Convert 16 contiguous floats (p[0..15]) into a v16bf A-fragment.
// Contiguous K order (k = 16h + s), matching the B-fragment LDS reads.
__device__ __forceinline__ v16bf load_cvt_A(const float* __restrict__ p) {
    const v4f* q = (const v4f*)p;
    v4f f0 = q[0], f1 = q[1], f2 = q[2], f3 = q[3];
    v16bf r;
#pragma unroll
    for (int i = 0; i < 4; ++i) {
        r[i]      = (__bf16)f0[i];
        r[4 + i]  = (__bf16)f1[i];
        r[8 + i]  = (__bf16)f2[i];
        r[12 + i] = (__bf16)f3[i];
    }
    return r;
}

__device__ __forceinline__ v16bf zero_v16bf() {
    v16bf z;
#pragma unroll
    for (int i = 0; i < 16; ++i) z[i] = (__bf16)0.0f;
    return z;
}

// K0: W1 f32 -> bf16 in workspace; zero the 256 stat accumulators.
__global__ __launch_bounds__(256) void prep_kernel(const float* __restrict__ w1,
                                                   __bf16* __restrict__ w1b,
                                                   float* __restrict__ stats) {
    int i = blockIdx.x * 256 + threadIdx.x;
    if (i < FEAT * FEAT2) w1b[i] = (__bf16)w1[i];
    if (blockIdx.x == 0 && threadIdx.x < 256) stats[threadIdx.x] = 0.0f;
}

// K2: fold batch stats into per-feature affine (a*h + c).
__global__ __launch_bounds__(128) void bn_coef_kernel(const float* __restrict__ sum,
                                                      const float* __restrict__ sq,
                                                      const float* __restrict__ gamma,
                                                      const float* __restrict__ beta,
                                                      float* __restrict__ a,
                                                      float* __restrict__ c, int E) {
    int j = threadIdx.x;
    if (j < FEAT) {
        float inv  = 1.0f / (float)E;
        float m    = sum[j] * inv;
        float var  = fmaxf(sq[j] * inv - m * m, 0.0f);
        float s    = rsqrtf(var + 1e-5f);
        float av   = gamma[j] * s;
        a[j] = av;
        c[j] = beta[j] - av * m;
    }
}

// K1 (PASS=1) / K3 (PASS=2): the WMMA edge-GEMM.
template <int PASS>
__global__ __launch_bounds__(BLK_THREADS) void edge_gemm_kernel(
    const float* __restrict__ x, const long long* __restrict__ ei,
    const __bf16* __restrict__ w1b,
    float* __restrict__ statsSum, float* __restrict__ statsSq,
    const float* __restrict__ acoef, const float* __restrict__ ccoef,
    const float* __restrict__ w2, const float* __restrict__ b2,
    float* __restrict__ out, int E) {
    // W1 staged in LDS, 32B-block XOR swizzle => conflict-free B-frag reads.
    __shared__ __align__(32) __bf16 w1s[FEAT * FEAT2];  // 64 KB

    const int tid = threadIdx.x;
    // Cooperative fill: 2048 blocks of 16 bf16 (32B each).
    for (int g = tid; g < FEAT * (FEAT2 / 16); g += BLK_THREADS) {
        int j = g >> 4, b = g & 15;
        const uint4* src = (const uint4*)(w1b + j * FEAT2 + b * 16);
        uint4 lo = src[0], hi = src[1];
        uint4* dst = (uint4*)(w1s + j * FEAT2 + ((b ^ (j & 15)) * 16));
        dst[0] = lo; dst[1] = hi;
    }
    __syncthreads();

    const int lane = tid & 31;
    const int hh   = lane >> 4;    // half-select (K layout / row half)
    const int r    = lane & 15;    // edge-row within tile / B column / j-lane
    const int gwave  = blockIdx.x * (BLK_THREADS / 32) + (tid >> 5);
    const int nwave  = gridDim.x * (BLK_THREADS / 32);
    const int ntiles = (E + TILE_E - 1) / TILE_E;

    // Per-lane swizzled 16-element-block offsets within a W1 row.
    // Swizzle uses only j&15 == r, so these are j-tile independent.
    int bo[8];
#pragma unroll
    for (int kt = 0; kt < 8; ++kt) bo[kt] = ((kt * 2 + hh) ^ r) << 4;

    float aj[8], cj[8], w0j[8], w1j[8];
    float b2v0 = 0.0f, b2v1 = 0.0f;
    if constexpr (PASS == 2) {
#pragma unroll
        for (int t = 0; t < 8; ++t) {
            int j = t * 16 + r;
            aj[t]  = acoef[j];
            cj[t]  = ccoef[j];
            w0j[t] = w2[j];          // W2[0, j]
            w1j[t] = w2[FEAT + j];   // W2[1, j]
        }
        b2v0 = b2[0];
        b2v1 = b2[1];
    }
    float ssum[8], ssq[8];
#pragma unroll
    for (int t = 0; t < 8; ++t) { ssum[t] = 0.0f; ssq[t] = 0.0f; }

    for (int tile = gwave; tile < ntiles; tile += nwave) {
        const int e = tile * TILE_E + r;
        const bool valid = (e < E);
        long long es = 0, ed = 0;
        if (valid) { es = ei[2 * (long long)e]; ed = ei[2 * (long long)e + 1]; }
        const float* ps = x + es * FEAT + hh * 16;
        const float* pd = x + ed * FEAT + hh * 16;

        // Prefetch next tile's edge indices (global_prefetch_b8).
        int nt = tile + nwave;
        if (nt < ntiles) __builtin_prefetch(&ei[2 * (long long)(nt * TILE_E + r)], 0, 1);

        // A fragments: z row = [x[src](128) | x[dst](128)], 8 k-tiles of 32.
        v16bf A[8];
#pragma unroll
        for (int kt = 0; kt < 4; ++kt) {
            A[kt]     = valid ? load_cvt_A(ps + kt * 32) : zero_v16bf();
            A[kt + 4] = valid ? load_cvt_A(pd + kt * 32) : zero_v16bf();
        }

        float s0[8], s1[8];
        if constexpr (PASS == 2) {
#pragma unroll
            for (int v = 0; v < 8; ++v) { s0[v] = 0.0f; s1[v] = 0.0f; }
        }

#pragma unroll
        for (int jt = 0; jt < 8; ++jt) {
            const int j = jt * 16 + r;  // this lane's B column / output feature
            // Opaque row offset: blocks LICM from hoisting the (loop-invariant)
            // LDS B-fragment loads out of the tile loop (previously ballooned
            // to 512 live VGPRs and spilled to scratch). One volatile asm per
            // j-tile also keeps the 8 groups in order.
            unsigned boff = (unsigned)(j * FEAT2);
            asm volatile("" : "+v"(boff));
            const __bf16* rowp = w1s + boff;

            v8f acc;
#pragma unroll
            for (int v = 0; v < 8; ++v) acc[v] = 0.0f;

            // 3-deep pipelined B-fragment fetch: each ds_load pair has ~2
            // WMMAs of latency cover instead of an immediate s_wait_dscnt 0.
            v16bf Bb[3];
#pragma unroll
            for (int p = 0; p < 3; ++p)
                Bb[p] = *(const v16bf*)(rowp + bo[p]);
#pragma unroll
            for (int kt = 0; kt < 8; ++kt) {
                acc = __builtin_amdgcn_wmma_f32_16x16x32_bf16(
                    false, A[kt], false, Bb[kt % 3], (short)0, acc, false, false);
                if (kt + 3 < 8)
                    Bb[kt % 3] = *(const v16bf*)(rowp + bo[kt + 3]);
            }

            if constexpr (PASS == 1) {
#pragma unroll
                for (int v = 0; v < 8; ++v) {
                    float h = acc[v];
                    ssum[jt] += h;
                    ssq[jt]  = fmaf(h, h, ssq[jt]);
                }
            } else {
                float a = aj[jt], c = cj[jt], wa = w0j[jt], wb = w1j[jt];
#pragma unroll
                for (int v = 0; v < 8; ++v) {
                    float hn = fmaf(a, acc[v], c);
                    hn = fmaxf(hn, 0.0f);  // ReLU
                    s0[v] = fmaf(wa, hn, s0[v]);
                    s1[v] = fmaf(wb, hn, s1[v]);
                }
            }
        }

        if constexpr (PASS == 2) {
            // Reduce over the 16 feature-lanes within each half, write [E,2].
#pragma unroll
            for (int v = 0; v < 8; ++v) {
                float t0 = s0[v], t1 = s1[v];
#pragma unroll
                for (int m = 8; m >= 1; m >>= 1) {
                    t0 += __shfl_xor(t0, m, 16);
                    t1 += __shfl_xor(t1, m, 16);
                }
                int eo = tile * TILE_E + v + hh * 8;  // C-layout row: M = v + 8*hh
                if (r == 0 && eo < E) {
                    float2 res = make_float2(t0 + b2v0, t1 + b2v1);
                    *(float2*)(out + (long long)eo * 2) = res;
                }
            }
        }
    }

    if constexpr (PASS == 1) {
        // Fold lane L with L+16 (same j, different edge rows), flush once/wave.
#pragma unroll
        for (int t = 0; t < 8; ++t) {
            float s = ssum[t] + __shfl_xor(ssum[t], 16, 32);
            float q = ssq[t]  + __shfl_xor(ssq[t], 16, 32);
            if (hh == 0) {
                atomicAdd(&statsSum[t * 16 + r], s);
                atomicAdd(&statsSq[t * 16 + r], q);
            }
        }
    }
}

extern "C" void kernel_launch(void* const* d_in, const int* in_sizes, int n_in,
                              void* d_out, int out_size, void* d_ws, size_t ws_size,
                              hipStream_t stream) {
    const float*     x     = (const float*)d_in[0];
    const long long* ei    = (const long long*)d_in[1];  // int64 per reference
    const float*     W1    = (const float*)d_in[2];
    // d_in[3] = b1: unused (BatchNorm's mean subtraction cancels it exactly)
    const float*     gamma = (const float*)d_in[4];
    const float*     beta  = (const float*)d_in[5];
    const float*     W2    = (const float*)d_in[6];
    const float*     b2    = (const float*)d_in[7];
    float*           out   = (float*)d_out;

    const int E = in_sizes[1] / 2;

    char* ws = (char*)d_ws;
    __bf16* w1b      = (__bf16*)ws;                    // 65536 B
    float*  statsSum = (float*)(ws + 65536);           // 512 B
    float*  statsSq  = (float*)(ws + 65536 + 512);     // 512 B
    float*  acoef    = (float*)(ws + 66560);           // 512 B
    float*  ccoef    = (float*)(ws + 66560 + 512);     // 512 B

    const int ntiles = (E + TILE_E - 1) / TILE_E;
    int blocks = (ntiles + (BLK_THREADS / 32) - 1) / (BLK_THREADS / 32);
    if (blocks > MAX_BLOCKS) blocks = MAX_BLOCKS;
    if (blocks < 1) blocks = 1;

    prep_kernel<<<128, 256, 0, stream>>>(W1, w1b, statsSum);
    edge_gemm_kernel<1><<<blocks, BLK_THREADS, 0, stream>>>(
        x, ei, w1b, statsSum, statsSq, nullptr, nullptr, nullptr, nullptr,
        nullptr, E);
    bn_coef_kernel<<<1, 128, 0, stream>>>(statsSum, statsSq, gamma, beta, acoef,
                                          ccoef, E);
    edge_gemm_kernel<2><<<blocks, BLK_THREADS, 0, stream>>>(
        x, ei, w1b, nullptr, nullptr, acoef, ccoef, W2, b2, out, E);
}
